// MultiHeadAttention_33517924778092
// MI455X (gfx1250) — compile-verified, matrix operand order checked by reading, not run on hardware
//
#include <hip/hip_runtime.h>
#include <hip/hip_bf16.h>
#include <stdint.h>

// ---------------------------------------------------------------------------
// MHA for MI455X (gfx1250): f16 WMMA pipeline, fp32 accumulate.
// GEMMs: 64x64 accumulator per wave (16 WMMAs / k-step).
// Attention: workgroup-cooperative K/V staging via async global->LDS
// (double buffered, ASYNCcnt + barrier), WMMA fragments from LDS.
// ---------------------------------------------------------------------------

#define B_   4
#define S_   2048
#define D_   1024
#define H_   16
#define HD_  64
#define BS_  (B_ * S_)          // 8192 rows

typedef __attribute__((ext_vector_type(16))) _Float16 v16h;
typedef __attribute__((ext_vector_type(8)))  _Float16 v8h;
typedef __attribute__((ext_vector_type(4)))  _Float16 v4h;
typedef __attribute__((ext_vector_type(8)))  float    v8f;
typedef __attribute__((ext_vector_type(4)))  int      v4i;

#if defined(__has_builtin)
#  if __has_builtin(__builtin_amdgcn_global_load_async_to_lds_b128)
#    define USE_ASYNC_LDS 1
#  endif
#endif
#ifndef USE_ASYNC_LDS
#  define USE_ASYNC_LDS 0
#endif

__device__ __forceinline__ v8f v8f_zero() {
  v8f z = {0.f, 0.f, 0.f, 0.f, 0.f, 0.f, 0.f, 0.f};
  return z;
}

__device__ __forceinline__ v8f wmma_f16(v16h a, v16h b, v8f c) {
  // D = A x B + C, 16x16x32 f16 -> f32
  return __builtin_amdgcn_wmma_f32_16x16x32_f16(
      /*neg_a=*/false, a, /*neg_b=*/false, b,
      /*c_mod=*/(short)0, c, /*reuse_a=*/false, /*reuse_b=*/false);
}

// Load a 16x32 f16 fragment in the CDNA5 WMMA A layout (B mirrors with N<->M).
// ISA 7.12.2: lane<16  holds row=lane,    K elems {0..7, 16..23}
//             lane>=16 holds row=lane-16, K elems {8..15, 24..31}
// p = tile origin (row0, k0); ld = row stride in halfs. Works for global & LDS.
__device__ __forceinline__ v16h load_frag(const _Float16* p, int ld) {
  const int lane = threadIdx.x & 31;
  const int r    = lane & 15;
  const int hh   = lane >> 4;
  const _Float16* q = p + (size_t)r * ld + hh * 8;
  v8h lo = *(const v8h*)(q);        // K {0..7} or {8..15}
  v8h hi = *(const v8h*)(q + 16);   // K {16..23} or {24..31}
  union { v16h v; v8h h[2]; } u;
  u.h[0] = lo;
  u.h[1] = hi;
  return u.v;
}

// 16B global -> LDS copy: async DMA (ASYNCcnt) when available, VGPR bounce else.
__device__ __forceinline__ void cp_b128(const _Float16* gsrc, _Float16* ldst) {
#if USE_ASYNC_LDS
  __builtin_amdgcn_global_load_async_to_lds_b128(
      (v4i*)gsrc, (v4i*)ldst, /*offset=*/0, /*cpol=*/0);
#else
  v8h t = *(const v8h*)gsrc;
  *(v8h*)ldst = t;
#endif
}

__device__ __forceinline__ void wait_async_le4() {
#if USE_ASYNC_LDS
  asm volatile("s_wait_asynccnt 0x4" ::: "memory");
#endif
}

// ---------------------------------------------------------------------------
// f32 -> f16 conversion (4-wide)
// ---------------------------------------------------------------------------
__global__ void cvt_f32_to_f16(const float* __restrict__ x,
                               _Float16* __restrict__ y, int n4) {
  int i = blockIdx.x * blockDim.x + threadIdx.x;
  if (i < n4) {
    float4 f = ((const float4*)x)[i];
    v4h h = {(_Float16)f.x, (_Float16)f.y, (_Float16)f.z, (_Float16)f.w};
    ((v4h*)y)[i] = h;
  }
}

// ---------------------------------------------------------------------------
// GEMM: Y = X @ W^T + bias.  X:[M,K] f16 rm, W:[N,K] f16 rm (torch Linear).
// Block = 128 threads (4 waves). Each wave: 64 rows x 64 cols (16 WMMA tiles,
// 128 accumulator VGPRs); K looped in steps of 32 -> 16 WMMAs : 16 B128 loads.
// MODE 0: Y f16 head-split  [B,H,S,HD]   (for Q, K)
// MODE 1: Y f16 transposed  [B,H,HD,S]   (for V -> Vt)
// MODE 2: Y f32 flat        [M, 1024]    (final output projection)
// ---------------------------------------------------------------------------
template <int MODE>
__global__ __launch_bounds__(128)
void gemm_xwT(const _Float16* __restrict__ X, const _Float16* __restrict__ W,
              const float* __restrict__ bias, void* __restrict__ Yv,
              int M, int K) {
  const int wave = threadIdx.x >> 5;
  const int lane = threadIdx.x & 31;
  const int r    = lane & 15;
  const int hh   = lane >> 4;
  const int m0   = blockIdx.x * 256 + wave * 64;   // four 16-row tiles per wave
  const int n0   = blockIdx.y * 64;                // four 16-col tiles

  v8f acc[4][4];
#pragma unroll
  for (int mt = 0; mt < 4; ++mt)
#pragma unroll
    for (int t = 0; t < 4; ++t) acc[mt][t] = v8f_zero();

  for (int k0 = 0; k0 < K; k0 += 32) {
    if (k0 + 32 < K) {
      __builtin_prefetch(X + (size_t)(m0 + r) * K + k0 + 32, 0, 1);
      __builtin_prefetch(W + (size_t)(n0 + r) * K + k0 + 32, 0, 1);
    }
    v16h a[4];
#pragma unroll
    for (int mt = 0; mt < 4; ++mt)
      a[mt] = load_frag(X + (size_t)(m0 + 16 * mt) * K + k0, K);
#pragma unroll
    for (int t = 0; t < 4; ++t) {
      v16h b = load_frag(W + (size_t)(n0 + 16 * t) * K + k0, K);
#pragma unroll
      for (int mt = 0; mt < 4; ++mt)
        acc[mt][t] = wmma_f16(a[mt], b, acc[mt][t]);
    }
  }

  // Epilogue: C layout element i -> (row = i + 8*hh, col = r) within a tile.
#pragma unroll
  for (int mt = 0; mt < 4; ++mt) {
#pragma unroll
    for (int t = 0; t < 4; ++t) {
      const int nn = n0 + 16 * t + r;
      const float bval = bias[nn];
#pragma unroll
      for (int i = 0; i < 8; ++i) {
        const int mm = m0 + 16 * mt + i + 8 * hh;
        const float val = acc[mt][t][i] + bval;
        if (MODE == 0) {
          const int b = mm >> 11, s = mm & (S_ - 1);
          const int h = nn >> 6,  d = nn & (HD_ - 1);
          ((_Float16*)Yv)[(((size_t)b * H_ + h) * S_ + s) * HD_ + d] =
              (_Float16)val;
        } else if (MODE == 1) {
          const int b = mm >> 11, s = mm & (S_ - 1);
          const int h = nn >> 6,  d = nn & (HD_ - 1);
          ((_Float16*)Yv)[(((size_t)b * H_ + h) * HD_ + d) * S_ + s] =
              (_Float16)val;
        } else {
          ((float*)Yv)[(size_t)mm * D_ + nn] = val;
        }
      }
    }
  }
}

// ---------------------------------------------------------------------------
// Causal flash attention. Workgroup = 4 waves = 64 query rows of one (b,h);
// wave w owns rows q0 = qbase + 16*w. All waves share the K/V stream, so K/V
// blocks (32 keys) are staged cooperatively into LDS with async global->LDS,
// double-buffered, with s_wait_asynccnt + barrier. WMMA fragments come from
// LDS (ds_load_b128). P transposed to A layout through per-wave LDS slab.
// Q:[B,H,S,HD] f16, K:[B,H,S,HD] f16, Vt:[B,H,HD,S] f16, O:[B,H,S,HD] f16.
// ---------------------------------------------------------------------------
#define KLD 72   // K-block LDS row stride (halfs); 144B, 16B aligned
#define VLD 40   // V-block / P LDS row stride (halfs); 80B, 16B aligned

__global__ __launch_bounds__(128)
void attn_flash(const _Float16* __restrict__ Q, const _Float16* __restrict__ Km,
                const _Float16* __restrict__ Vt, _Float16* __restrict__ O) {
  __shared__ __align__(16) _Float16 KlA[2][32 * KLD];  // 32 keys x 64 dims
  __shared__ __align__(16) _Float16 VlA[2][64 * VLD];  // 64 dims x 32 keys
  __shared__ __align__(16) _Float16 plds[4][16 * VLD]; // per-wave P tile

  const int tid  = threadIdx.x;
  const int wave = tid >> 5;
  const int lane = tid & 31;
  const int r    = lane & 15;
  const int hh   = lane >> 4;

  const int tiles_per_bh = S_ / 64;
  const int bh    = blockIdx.x / tiles_per_bh;
  const int qbase = (blockIdx.x % tiles_per_bh) * 64;
  const int q0    = qbase + wave * 16;

  const _Float16* Qh = Q  + (size_t)bh * S_ * HD_;
  const _Float16* Kh = Km + (size_t)bh * S_ * HD_;
  const _Float16* Vh = Vt + (size_t)bh * HD_ * S_;
  _Float16* pl = &plds[wave][0];

  const v16h aq0 = load_frag(Qh + (size_t)q0 * HD_ + 0,  HD_);
  const v16h aq1 = load_frag(Qh + (size_t)q0 * HD_ + 32, HD_);

  v8f o[4];
#pragma unroll
  for (int t = 0; t < 4; ++t) o[t] = v8f_zero();
  float rm[8], rl[8];
#pragma unroll
  for (int i = 0; i < 8; ++i) { rm[i] = -1e30f; rl[i] = 0.f; }

  const float SCALE = 0.125f;        // 1/sqrt(HD)
  const float L2E   = 1.44269504f;   // log2(e)
  // causal: block-uniform #key-blocks covering row qbase+63
  const int njmax = (qbase + 95) >> 5;

  // stage K/V block j into LDS buffer: 4 x cp_b128 per thread (8KB total)
  auto stage = [&](int kb, int buf) {
    {  // K: 32 rows x 64 halfs; 4 threads/row, 16 halfs each
      const int rr = tid >> 2, c = tid & 3;
      const _Float16* src = Kh + (size_t)(kb + rr) * HD_ + c * 16;
      _Float16* dst = &KlA[buf][rr * KLD + c * 16];
      cp_b128(src, dst);
      cp_b128(src + 8, dst + 8);
    }
    {  // Vt: 64 rows x 32 halfs; 2 threads/row, 16 halfs each
      const int d = tid >> 1, c = tid & 1;
      const _Float16* src = Vh + (size_t)d * S_ + kb + c * 16;
      _Float16* dst = &VlA[buf][d * VLD + c * 16];
      cp_b128(src, dst);
      cp_b128(src + 8, dst + 8);
    }
  };

  stage(0, 0);

  for (int j = 0; j < njmax; ++j) {
    const int kb = j * 32;
    if (j + 1 < njmax) stage(kb + 32, (j + 1) & 1);
    // oldest 4 async ops (= this buffer's staging) retired, then all waves sync
    wait_async_le4();
    __syncthreads();

    if (kb <= q0 + 15) {  // wave-uniform causal skip
      const _Float16* Kl = &KlA[j & 1][0];
      const _Float16* Vl = &VlA[j & 1][0];

      // ---- scores: S = (Q @ K^T) * 1/8 for 32 keys (fragments from LDS) ---
      v16h bk00 = load_frag(Kl + 0 * KLD + 0,  KLD);
      v16h bk01 = load_frag(Kl + 0 * KLD + 32, KLD);
      v16h bk10 = load_frag(Kl + 16 * KLD + 0,  KLD);
      v16h bk11 = load_frag(Kl + 16 * KLD + 32, KLD);
      v8f s0 = wmma_f16(aq0, bk00, wmma_f16(aq1, bk01, v8f_zero()));
      v8f s1 = wmma_f16(aq0, bk10, wmma_f16(aq1, bk11, v8f_zero()));

      // ---- causal mask + online softmax (per C-layout element) -----------
#pragma unroll
      for (int i = 0; i < 8; ++i) {
        const int mm = q0 + i + 8 * hh;                 // global query row
        float x0 = (kb + r      <= mm) ? s0[i] * SCALE : -1e30f;
        float x1 = (kb + 16 + r <= mm) ? s1[i] * SCALE : -1e30f;
        float mx = fmaxf(x0, x1);
        mx = fmaxf(mx, __shfl_xor(mx, 1, 32));          // stays within the
        mx = fmaxf(mx, __shfl_xor(mx, 2, 32));          // 16-lane half on
        mx = fmaxf(mx, __shfl_xor(mx, 4, 32));          // wave32
        mx = fmaxf(mx, __shfl_xor(mx, 8, 32));
        const float mnew = fmaxf(rm[i], mx);
        const float corr = exp2f((rm[i] - mnew) * L2E);
        rm[i] = mnew;
        const float p0 = exp2f((x0 - mnew) * L2E);
        const float p1 = exp2f((x1 - mnew) * L2E);
        rl[i] = rl[i] * corr + p0 + p1;                 // lane-partial row sum
        o[0][i] *= corr; o[1][i] *= corr; o[2][i] *= corr; o[3][i] *= corr;
        // stage P (16x32) row-major into per-wave slab for A-layout reload
        pl[(i + 8 * hh) * VLD + r]      = (_Float16)p0;
        pl[(i + 8 * hh) * VLD + 16 + r] = (_Float16)p1;
      }
      // same-wave DS ordering guarantees store->load visibility
      v16h pa = load_frag(pl, VLD);

      // ---- O += P @ V (B operand = rows of staged Vt block) --------------
#pragma unroll
      for (int t = 0; t < 4; ++t) {
        v16h bv = load_frag(Vl + (size_t)(16 * t) * VLD, VLD);
        o[t] = wmma_f16(pa, bv, o[t]);
      }
    }
    __syncthreads();  // all waves done with buf[j&1] before it is re-staged
  }

  // ---- finalize: reduce row sums across the 16-lane half, divide, store ---
#pragma unroll
  for (int i = 0; i < 8; ++i) {
    float l = rl[i];
    l += __shfl_xor(l, 1, 32);
    l += __shfl_xor(l, 2, 32);
    l += __shfl_xor(l, 4, 32);
    l += __shfl_xor(l, 8, 32);
    const float inv = 1.0f / l;
    const int mm = q0 + i + 8 * hh;
#pragma unroll
    for (int t = 0; t < 4; ++t) {
      O[((size_t)bh * S_ + mm) * HD_ + 16 * t + r] = (_Float16)(o[t][i] * inv);
    }
  }
}

// ---------------------------------------------------------------------------
// Host-side orchestration
// ---------------------------------------------------------------------------
extern "C" void kernel_launch(void* const* d_in, const int* in_sizes, int n_in,
                              void* d_out, int out_size, void* d_ws,
                              size_t ws_size, hipStream_t stream) {
  (void)in_sizes; (void)n_in; (void)out_size; (void)ws_size;
  const float* q    = (const float*)d_in[0];
  const float* k    = (const float*)d_in[1];
  const float* v    = (const float*)d_in[2];
  /* d_in[3] = mask: causal tril, reproduced analytically in-kernel */
  const float* wq   = (const float*)d_in[4];
  const float* bq   = (const float*)d_in[5];
  const float* wk   = (const float*)d_in[6];
  const float* bk   = (const float*)d_in[7];
  const float* wv   = (const float*)d_in[8];
  const float* bv   = (const float*)d_in[9];
  const float* wo   = (const float*)d_in[10];
  const float* bo   = (const float*)d_in[11];

  const size_t nQKV = (size_t)BS_ * D_;   // 8388608 elems
  const size_t nW   = (size_t)D_ * D_;    // 1048576 elems

  _Float16* w   = (_Float16*)d_ws;
  _Float16* qf  = w;            // f16 inputs
  _Float16* kf  = qf  + nQKV;
  _Float16* vf  = kf  + nQKV;
  _Float16* wqf = vf  + nQKV;   // f16 weights
  _Float16* wkf = wqf + nW;
  _Float16* wvf = wkf + nW;
  _Float16* wof = wvf + nW;
  _Float16* Qb  = wof + nW;     // projected Q [B,H,S,HD]
  _Float16* Kb  = Qb  + nQKV;   // projected K [B,H,S,HD]
  _Float16* Vtb = Kb  + nQKV;   // projected V^T [B,H,HD,S]
  _Float16* Xa  = Vtb + nQKV;   // attention out, flat [B,H,S,HD]

  // 1) f32 -> f16 conversions
  {
    const int n4q = (int)(nQKV / 4), n4w = (int)(nW / 4);
    const int tb = 256;
    cvt_f32_to_f16<<<(n4q + tb - 1) / tb, tb, 0, stream>>>(q,  qf,  n4q);
    cvt_f32_to_f16<<<(n4q + tb - 1) / tb, tb, 0, stream>>>(k,  kf,  n4q);
    cvt_f32_to_f16<<<(n4q + tb - 1) / tb, tb, 0, stream>>>(v,  vf,  n4q);
    cvt_f32_to_f16<<<(n4w + tb - 1) / tb, tb, 0, stream>>>(wq, wqf, n4w);
    cvt_f32_to_f16<<<(n4w + tb - 1) / tb, tb, 0, stream>>>(wk, wkf, n4w);
    cvt_f32_to_f16<<<(n4w + tb - 1) / tb, tb, 0, stream>>>(wv, wvf, n4w);
    cvt_f32_to_f16<<<(n4w + tb - 1) / tb, tb, 0, stream>>>(wo, wof, n4w);
  }

  // 2) QKV projections (WMMA)
  dim3 gg(BS_ / 256, D_ / 64);
  gemm_xwT<0><<<gg, 128, 0, stream>>>(qf, wqf, bq, (void*)Qb,  BS_, D_);
  gemm_xwT<0><<<gg, 128, 0, stream>>>(kf, wkf, bk, (void*)Kb,  BS_, D_);
  gemm_xwT<1><<<gg, 128, 0, stream>>>(vf, wvf, bv, (void*)Vtb, BS_, D_);

  // 3) causal flash attention (WMMA + async K/V staging)
  attn_flash<<<B_ * H_ * (S_ / 64), 128, 0, stream>>>(Qb, Kb, Vtb, Xa);

  // 4) output projection -> f32 (WMMA). Feeding the flat [B,H,S,HD] buffer as
  //    rows of 1024 reproduces the reference's no-transpose reshape exactly.
  gemm_xwT<2><<<gg, 128, 0, stream>>>(Xa, wof, bo, d_out, BS_, D_);
}
